// ConditionalDecoder_78872779423833
// MI455X (gfx1250) — matrix operand order repaired and checked by reading, not compiled, not run on hardware
//
#include <hip/hip_runtime.h>
#include <hip/hip_bf16.h>
#include <cstddef>

// ---------------------------------------------------------------------------
// CDNA5 (gfx1250) WMMA helpers: D = A(16x32 f16) * B(32x16 f16) + C(16x16 f32)
// Layouts per CDNA5 ISA 7.12.2 (wave32).
// ---------------------------------------------------------------------------
typedef __attribute__((ext_vector_type(16))) _Float16 v16h;
typedef __attribute__((ext_vector_type(8)))  float    v8f;
typedef __attribute__((ext_vector_type(8)))  _Float16 h8;   // 16B of f16

__device__ __forceinline__ v8f wmma_f16(v16h a, v16h b, v8f c) {
  return __builtin_amdgcn_wmma_f32_16x16x32_f16(
      /*neg_a=*/false, a, /*neg_b=*/false, b,
      /*c_mod=*/(short)0, c, /*reuse_a=*/false, /*reuse_b=*/false);
}

// A fragment from an f16 row-major LDS tile (16 rows, stride lda halfs), K-chunk 32.
// lanes 0-15: M=lane, VGPR0..3 K=0..7, VGPR4..7 K=16..23
// lanes 16-31: M=lane-16, VGPR0..3 K=8..15, VGPR4..7 K=24..31
__device__ __forceinline__ v16h load_a_frag(const _Float16* A, int lda, int lane) {
  int m  = lane & 15;
  int kh = ((lane >> 4) & 1) * 8;
  v16h a;
#pragma unroll
  for (int v = 0; v < 8; ++v) {
    int k = ((v < 4) ? 0 : 16) + kh + ((v & 3) << 1);
    a[2 * v]     = A[m * lda + k];
    a[2 * v + 1] = A[m * lda + k + 1];
  }
  return a;
}

// B fragment from pre-converted, pre-padded f16 weights W[K][NC] row-major.
// K multiple of 32 and NC multiple of 16 (enforced by padding at conversion),
// so all bounds checks fold away: pure global f16 loads, no selects.
template <int K, int NC>
__device__ __forceinline__ v16h load_b_frag_h(const _Float16* __restrict__ W,
                                              int k0, int n0, int lane) {
  static_assert(K % 32 == 0 && NC % 16 == 0, "padded dims required");
  int n  = n0 + (lane & 15);
  int kb = k0 + ((lane >> 4) & 1) * 8;
  v16h b;
#pragma unroll
  for (int v = 0; v < 8; ++v) {
    int k = kb + ((v < 4) ? 0 : 16) + ((v & 3) << 1);
    b[2 * v]     = W[(size_t)k * NC + n];
    b[2 * v + 1] = W[(size_t)(k + 1) * NC + n];
  }
  return b;
}

// ---------------------------------------------------------------------------
// Templated WMMA GEMM: C[M,NC] = act(concat(A0,A1)[M,K0+K1] @ W + bias)
// A (f32) staged to LDS f16 with float4 loads; optional row-gather via LDS.
// W is pre-converted f16 padded to [K][NCP]. NC = logical output width.
// 1 wave/block; blockIdx.{x,y} = {M,N}-tile. ACT: 0=none, 1=relu, 2=sigmoid
// ---------------------------------------------------------------------------
template <int K0, int K1, int NC, int NCP, int ACT, bool GATHER>
__global__ void __launch_bounds__(32)
k_gemm_t(const float* __restrict__ A0, const float* __restrict__ A1,
         const int* __restrict__ rowidx,
         const _Float16* __restrict__ W, const float* __restrict__ bias,
         float* __restrict__ C) {
  constexpr int K   = K0 + K1;
  constexpr int Q0  = K0 / 4;
  constexpr int Q1  = (K1 > 0) ? (K1 / 4) : 1;
  constexpr int NB1 = (K1 > 0) ? 16 * (K1 / 4) : 0;
  static_assert(K % 32 == 0, "K multiple of 32");
  __shared__ _Float16 sA[16][K];
  __shared__ int sRow[16];
  int lane = threadIdx.x;
  int row0 = blockIdx.x * 16;
  int n0   = blockIdx.y * 16;

  __builtin_prefetch(W + (size_t)n0, 0, 1);

  if (GATHER && lane < 16) sRow[lane] = rowidx[row0 + lane];
  __syncthreads();

  for (int i = lane; i < 16 * Q0; i += 32) {
    int r = i / Q0, q = i % Q0;
    int rr = GATHER ? sRow[r] : (row0 + r);
    float4 v = *(const float4*)(A0 + (size_t)rr * K0 + q * 4);
    sA[r][q * 4 + 0] = (_Float16)v.x;
    sA[r][q * 4 + 1] = (_Float16)v.y;
    sA[r][q * 4 + 2] = (_Float16)v.z;
    sA[r][q * 4 + 3] = (_Float16)v.w;
  }
  if (K1 > 0) {
    for (int i = lane; i < NB1; i += 32) {
      int r = i / Q1, q = i % Q1;
      float4 v = *(const float4*)(A1 + (size_t)(row0 + r) * K1 + q * 4);
      sA[r][K0 + q * 4 + 0] = (_Float16)v.x;
      sA[r][K0 + q * 4 + 1] = (_Float16)v.y;
      sA[r][K0 + q * 4 + 2] = (_Float16)v.z;
      sA[r][K0 + q * 4 + 3] = (_Float16)v.w;
    }
  }
  __syncthreads();

  v8f acc = {};
#pragma unroll
  for (int kt = 0; kt < K / 32; ++kt) {
    v16h a = load_a_frag(&sA[0][kt * 32], K, lane);
    v16h b = load_b_frag_h<K, NCP>(W, kt * 32, n0, lane);
    acc = wmma_f16(a, b, acc);
  }

  int n  = n0 + (lane & 15);
  int mb = ((lane >> 4) & 1) * 8;
  float bb = bias[(n < NC) ? n : (NC - 1)];
#pragma unroll
  for (int r = 0; r < 8; ++r) {
    float v = acc[r] + bb;
    if (ACT == 1) v = fmaxf(v, 0.f);
    if (ACT == 2) v = 1.f / (1.f + __expf(-v));
    if (n < NC) C[(size_t)(row0 + mb + r) * NC + n] = v;
  }
}

// ---------------------------------------------------------------------------
// Fused edge heads: per 16-edge tile builds ei=[he16[src],he16[dst]] (f16 LDS,
// b128 loads, zero conversion), runs ee (128->64->1 sigmoid), dd (128->64->1
// sigmoid), ef (128->64->4 softmax), ep (128->128->128->4), all WMMA with
// intermediates in LDS. (em* MLP is dead code in the reference and skipped.)
// Narrow output weights are padded to 16 columns -> no selects anywhere.
// ---------------------------------------------------------------------------
__device__ __forceinline__ void head_128to64_relu(const _Float16* sIn,
                                                  const _Float16* W1, const float* b1,
                                                  _Float16* sOut, int lane) {
  for (int nt = 0; nt < 4; ++nt) {
    v8f acc = {};
#pragma unroll
    for (int kt = 0; kt < 4; ++kt) {
      v16h a = load_a_frag(sIn + kt * 32, 128, lane);
      v16h b = load_b_frag_h<128, 64>(W1, kt * 32, nt * 16, lane);
      acc = wmma_f16(a, b, acc);
    }
    int n = lane & 15, mb = ((lane >> 4) & 1) * 8;
    float bb = b1[nt * 16 + n];
#pragma unroll
    for (int r = 0; r < 8; ++r)
      sOut[(mb + r) * 128 + nt * 16 + n] = (_Float16)fmaxf(acc[r] + bb, 0.f);
  }
}

// second layer of a head, 64 -> (<=16, padded weights)
__device__ __forceinline__ v8f head_64toVec(const _Float16* sU, const _Float16* W2,
                                            int lane) {
  v8f acc = {};
#pragma unroll
  for (int kt = 0; kt < 2; ++kt) {
    v16h a = load_a_frag(sU + kt * 32, 128, lane);
    v16h b = load_b_frag_h<64, 16>(W2, kt * 32, 0, lane);
    acc = wmma_f16(a, b, acc);
  }
  return acc;
}

__global__ void __launch_bounds__(32)
k_edge_heads(const _Float16* __restrict__ he16,
             const int* __restrict__ src, const int* __restrict__ dst,
             const _Float16* ee1w, const float* ee1b,
             const _Float16* ee2w, const float* ee2b,
             const _Float16* dd1w, const float* dd1b,
             const _Float16* dd2w, const float* dd2b,
             const _Float16* ef1w, const float* ef1b,
             const _Float16* ef2w, const float* ef2b,
             const _Float16* ep1w, const float* ep1b,
             const _Float16* ep2w, const float* ep2b,
             const _Float16* ep3w, const float* ep3b,
             float* __restrict__ out_exist,
             float* __restrict__ out_dist,
             float* __restrict__ out_efeat,
             float* __restrict__ edge_attr, int E) {
  __shared__ _Float16 sEI[16][128];
  __shared__ _Float16 sT1[16][128];
  __shared__ _Float16 sT2[16][128];
  __shared__ float sO[16][16];
  __shared__ int sIdx[2][16];
  int lane = threadIdx.x;
  int e0 = blockIdx.x * 16;

  __builtin_prefetch(ep1w, 0, 1);
  __builtin_prefetch(ep2w, 0, 1);

  {
    const int* p = (lane < 16) ? src : dst;
    sIdx[(lane >> 4) & 1][lane & 15] = p[e0 + (lane & 15)];
  }
  __syncthreads();

  // stage ei = [he16[src], he16[dst]]: 16B f16 vector loads, no conversion
  for (int i = lane; i < 16 * 16; i += 32) {
    int r = i >> 4, q = i & 15;
    int node = sIdx[q >> 3][r];
    h8 v = *(const h8*)(he16 + (size_t)node * 64 + (q & 7) * 8);
    *(h8*)&sEI[r][q * 8] = v;
  }
  __syncthreads();

  // --- ep: 128 -> 128 -> 128 -> 4  (edge_attr) ---
  for (int nt = 0; nt < 8; ++nt) {
    v8f acc = {};
#pragma unroll
    for (int kt = 0; kt < 4; ++kt) {
      v16h a = load_a_frag(&sEI[0][kt * 32], 128, lane);
      v16h b = load_b_frag_h<128, 128>(ep1w, kt * 32, nt * 16, lane);
      acc = wmma_f16(a, b, acc);
    }
    int n = lane & 15, mb = ((lane >> 4) & 1) * 8;
    float bb = ep1b[nt * 16 + n];
#pragma unroll
    for (int r = 0; r < 8; ++r)
      sT1[mb + r][nt * 16 + n] = (_Float16)fmaxf(acc[r] + bb, 0.f);
  }
  __syncthreads();
  for (int nt = 0; nt < 8; ++nt) {
    v8f acc = {};
#pragma unroll
    for (int kt = 0; kt < 4; ++kt) {
      v16h a = load_a_frag(&sT1[0][kt * 32], 128, lane);
      v16h b = load_b_frag_h<128, 128>(ep2w, kt * 32, nt * 16, lane);
      acc = wmma_f16(a, b, acc);
    }
    int n = lane & 15, mb = ((lane >> 4) & 1) * 8;
    float bb = ep2b[nt * 16 + n];
#pragma unroll
    for (int r = 0; r < 8; ++r)
      sT2[mb + r][nt * 16 + n] = (_Float16)fmaxf(acc[r] + bb, 0.f);
  }
  __syncthreads();
  {
    v8f acc = {};
#pragma unroll
    for (int kt = 0; kt < 4; ++kt) {
      v16h a = load_a_frag(&sT2[0][kt * 32], 128, lane);
      v16h b = load_b_frag_h<128, 16>(ep3w, kt * 32, 0, lane);
      acc = wmma_f16(a, b, acc);
    }
    int n = lane & 15, mb = ((lane >> 4) & 1) * 8;
    float bb = ep3b[(n < 4) ? n : 3];
#pragma unroll
    for (int r = 0; r < 8; ++r)
      if (n < 4) edge_attr[(size_t)(e0 + mb + r) * 4 + n] = acc[r] + bb;
  }
  __syncthreads();

  // --- ee: 128 -> 64 -> 1 sigmoid (edge_existence) ---
  head_128to64_relu(&sEI[0][0], ee1w, ee1b, &sT1[0][0], lane);
  __syncthreads();
  {
    v8f o = head_64toVec(&sT1[0][0], ee2w, lane);
    int n = lane & 15, mb = ((lane >> 4) & 1) * 8;
    if (n == 0)
#pragma unroll
      for (int r = 0; r < 8; ++r)
        out_exist[e0 + mb + r] = 1.f / (1.f + __expf(-(o[r] + ee2b[0])));
  }
  __syncthreads();

  // --- dd: 128 -> 64 -> 1 sigmoid (distances) ---
  head_128to64_relu(&sEI[0][0], dd1w, dd1b, &sT1[0][0], lane);
  __syncthreads();
  {
    v8f o = head_64toVec(&sT1[0][0], dd2w, lane);
    int n = lane & 15, mb = ((lane >> 4) & 1) * 8;
    if (n == 0)
#pragma unroll
      for (int r = 0; r < 8; ++r)
        out_dist[e0 + mb + r] = 1.f / (1.f + __expf(-(o[r] + dd2b[0])));
  }
  __syncthreads();

  // --- ef: 128 -> 64 -> 4 softmax (edge_feats) ---
  head_128to64_relu(&sEI[0][0], ef1w, ef1b, &sT1[0][0], lane);
  __syncthreads();
  {
    v8f o = head_64toVec(&sT1[0][0], ef2w, lane);
    int n = lane & 15, mb = ((lane >> 4) & 1) * 8;
    float bb = (n < 4) ? ef2b[n] : 0.f;
#pragma unroll
    for (int r = 0; r < 8; ++r) sO[mb + r][n] = o[r] + bb;
  }
  __syncthreads();
  if (lane < 16) {
    float mx = -1e30f;
    for (int j = 0; j < 4; ++j) mx = fmaxf(mx, sO[lane][j]);
    float s = 0.f, e_[4];
    for (int j = 0; j < 4; ++j) { e_[j] = __expf(sO[lane][j] - mx); s += e_[j]; }
    float inv = 1.f / s;
    for (int j = 0; j < 4; ++j)
      out_efeat[(size_t)(e0 + lane) * 4 + j] = e_[j] * inv;
  }
}

// ---------------------------------------------------------------------------
// Fused EGNN edge step: m = relu([he_s,he_d,d2,ea]@Wm) via WMMA (K=133, both
// A tile and weights zero-padded to 160 -> no guards), s = m@xw, scatter-add
// m into agg and rel*s into posdelta.
// ---------------------------------------------------------------------------
__global__ void __launch_bounds__(32)
k_egnn_edge(const _Float16* __restrict__ he16, const float* __restrict__ pos,
            const int* __restrict__ src, const int* __restrict__ dst,
            const float* __restrict__ eattr,
            const _Float16* __restrict__ msgw, const float* __restrict__ msgb,
            const float* __restrict__ xww, const float* __restrict__ xwb,
            float* __restrict__ agg, float* __restrict__ posdelta, int E) {
  __shared__ _Float16 sA[16][160];
  __shared__ float sM[16][64];
  __shared__ float sRel[16][3];
  __shared__ float sS[16];
  __shared__ int sIdx[2][16];
  int lane = threadIdx.x;
  int e0 = blockIdx.x * 16;

  __builtin_prefetch(msgw, 0, 1);

  {
    const int* p = (lane < 16) ? src : dst;
    sIdx[(lane >> 4) & 1][lane & 15] = p[e0 + (lane & 15)];
  }
  __syncthreads();

  if (lane < 16) {
    int a = sIdx[0][lane], b = sIdx[1][lane];
    float rx = pos[(size_t)a * 3 + 0] - pos[(size_t)b * 3 + 0];
    float ry = pos[(size_t)a * 3 + 1] - pos[(size_t)b * 3 + 1];
    float rz = pos[(size_t)a * 3 + 2] - pos[(size_t)b * 3 + 2];
    sRel[lane][0] = rx; sRel[lane][1] = ry; sRel[lane][2] = rz;
    sA[lane][128] = (_Float16)(rx * rx + ry * ry + rz * rz);
    float4 ea = *(const float4*)(eattr + (size_t)(e0 + lane) * 4);
    sA[lane][129] = (_Float16)ea.x;
    sA[lane][130] = (_Float16)ea.y;
    sA[lane][131] = (_Float16)ea.z;
    sA[lane][132] = (_Float16)ea.w;
#pragma unroll
    for (int k = 133; k < 160; ++k) sA[lane][k] = (_Float16)0.f;
  }
  for (int i = lane; i < 16 * 16; i += 32) {
    int r = i >> 4, q = i & 15;
    int node = sIdx[q >> 3][r];
    h8 v = *(const h8*)(he16 + (size_t)node * 64 + (q & 7) * 8);
    *(h8*)&sA[r][q * 8] = v;
  }
  __syncthreads();

  for (int nt = 0; nt < 4; ++nt) {
    v8f acc = {};
#pragma unroll
    for (int kt = 0; kt < 5; ++kt) {
      v16h a = load_a_frag(&sA[0][kt * 32], 160, lane);
      v16h b = load_b_frag_h<160, 64>(msgw, kt * 32, nt * 16, lane);
      acc = wmma_f16(a, b, acc);
    }
    int n = lane & 15, mb = ((lane >> 4) & 1) * 8;
    float bb = msgb[nt * 16 + n];
#pragma unroll
    for (int r = 0; r < 8; ++r)
      sM[mb + r][nt * 16 + n] = fmaxf(acc[r] + bb, 0.f);
  }
  __syncthreads();
  if (lane < 16) {
    float s = xwb[0];
    for (int c = 0; c < 64; ++c) s += sM[lane][c] * xww[c];
    sS[lane] = s;
  }
  __syncthreads();
  for (int i = lane; i < 16 * 64; i += 32) {
    int r = i >> 6, c = i & 63;
    atomicAdd(&agg[(size_t)sIdx[1][r] * 64 + c], sM[r][c]);
  }
  for (int i = lane; i < 48; i += 32) {
    int r = i / 3, d = i % 3;
    atomicAdd(&posdelta[(size_t)sIdx[1][r] * 3 + d], sRel[r][d] * sS[r]);
  }
}

// ---------------------------------------------------------------------------
// Small elementwise / reduction kernels
// ---------------------------------------------------------------------------
__global__ void k_cvt_f16(const float* __restrict__ in, _Float16* __restrict__ out,
                          int n) {
  int i = blockIdx.x * 256 + threadIdx.x;
  if (i < n) out[i] = (_Float16)in[i];
}

// convert W[K][NC] f32 -> padded f16 [KP][NCP] with zero fill
__global__ void k_cvt_pad(const float* __restrict__ in, _Float16* __restrict__ out,
                          int K, int NC, int KP, int NCP) {
  int i = blockIdx.x * 256 + threadIdx.x;
  if (i >= KP * NCP) return;
  int k = i / NCP, n = i % NCP;
  float v = (k < K && n < NC) ? in[k * NC + n] : 0.f;
  out[i] = (_Float16)v;
}

__global__ void k_latent(const float* __restrict__ z, const float* __restrict__ tp,
                         const float* __restrict__ w, const float* __restrict__ b,
                         float* __restrict__ h, int B) {
  int i = blockIdx.x * 256 + threadIdx.x;
  if (i >= B * 64) return;
  int r = i / 64, c = i % 64;
  float acc = b[c];
  for (int k = 0; k < 32; ++k) acc += z[r * 32 + k] * w[k * 64 + c];
  acc += tp[r] * w[32 * 64 + c];
  h[i] = acc;
}

__global__ void k_bn_stats(const float* __restrict__ X, int M, int C, int relu_in,
                           float* __restrict__ mean, float* __restrict__ var) {
  int c = blockIdx.x;
  __shared__ float ss[256], ss2[256];
  float s = 0.f, s2 = 0.f;
  for (int m = threadIdx.x; m < M; m += 256) {
    float v = X[(size_t)m * C + c];
    if (relu_in) v = fmaxf(v, 0.f);
    s += v; s2 += v * v;
  }
  ss[threadIdx.x] = s; ss2[threadIdx.x] = s2;
  __syncthreads();
  for (int o = 128; o > 0; o >>= 1) {
    if (threadIdx.x < o) { ss[threadIdx.x] += ss[threadIdx.x + o]; ss2[threadIdx.x] += ss2[threadIdx.x + o]; }
    __syncthreads();
  }
  if (threadIdx.x == 0) {
    float mu = ss[0] / M;
    mean[c] = mu;
    var[c] = ss2[0] / M - mu * mu;
  }
}

__global__ void k_bn_apply(float* __restrict__ X, int M, int C, int relu_in, int relu_out,
                           const float* __restrict__ mean, const float* __restrict__ var,
                           const float* __restrict__ g, const float* __restrict__ b) {
  size_t i = (size_t)blockIdx.x * 256 + threadIdx.x;
  if (i >= (size_t)M * C) return;
  int c = (int)(i % C);
  float v = X[i];
  if (relu_in) v = fmaxf(v, 0.f);
  v = g[c] * (v - mean[c]) * rsqrtf(var[c] + 1e-5f) + b[c];
  if (relu_out) v = fmaxf(v, 0.f);
  X[i] = v;
}

__global__ void k_softmax_rows(float* __restrict__ X, int M, int C) {
  int r = blockIdx.x * 256 + threadIdx.x;
  if (r >= M) return;
  float mx = -1e30f;
  for (int c = 0; c < C; ++c) mx = fmaxf(mx, X[(size_t)r * C + c]);
  float s = 0.f;
  for (int c = 0; c < C; ++c) { float e = __expf(X[(size_t)r * C + c] - mx); X[(size_t)r * C + c] = e; s += e; }
  float inv = 1.f / s;
  for (int c = 0; c < C; ++c) X[(size_t)r * C + c] *= inv;
}

__global__ void k_posnorm(const float* __restrict__ pin, float* __restrict__ pout, int N) {
  int n = blockIdx.x * 256 + threadIdx.x;
  if (n >= N) return;
  float x = pin[(size_t)n * 3], y = pin[(size_t)n * 3 + 1], z = pin[(size_t)n * 3 + 2];
  float inv = 1.f / (sqrtf(x * x + y * y + z * z) + 1e-12f);
  pout[(size_t)n * 3] = x * inv; pout[(size_t)n * 3 + 1] = y * inv; pout[(size_t)n * 3 + 2] = z * inv;
}

__global__ void k_fill(float* __restrict__ p, size_t n, float v) {
  size_t i = (size_t)blockIdx.x * 256 + threadIdx.x;
  if (i < n) p[i] = v;
}

__global__ void k_deg(const int* __restrict__ dst, float* __restrict__ deg, int E) {
  int e = blockIdx.x * 256 + threadIdx.x;
  if (e < E) atomicAdd(&deg[dst[e]], 1.f);
}

// he += h_upd, and refresh the f16 mirror used by the edge kernels
__global__ void k_axpy_dual(float* __restrict__ y, _Float16* __restrict__ y16,
                            const float* __restrict__ x, size_t n) {
  size_t i = (size_t)blockIdx.x * 256 + threadIdx.x;
  if (i < n) {
    float v = y[i] + x[i];
    y[i] = v;
    y16[i] = (_Float16)v;
  }
}

__global__ void k_pos_update(float* __restrict__ pos, const float* __restrict__ pd,
                             const float* __restrict__ deg, float coef, int N) {
  int n = blockIdx.x * 256 + threadIdx.x;
  if (n >= N) return;
  float inv = coef / (deg[n] + 1e-6f);
  pos[(size_t)n * 3] += pd[(size_t)n * 3] * inv;
  pos[(size_t)n * 3 + 1] += pd[(size_t)n * 3 + 1] * inv;
  pos[(size_t)n * 3 + 2] += pd[(size_t)n * 3 + 2] * inv;
}

__global__ void k_directions(const float* __restrict__ pos, const int* __restrict__ src,
                             const int* __restrict__ dst, float* __restrict__ out, int E) {
  int e = blockIdx.x * 256 + threadIdx.x;
  if (e >= E) return;
  int a = src[e], b = dst[e];
  float x = pos[(size_t)b * 3] - pos[(size_t)a * 3];
  float y = pos[(size_t)b * 3 + 1] - pos[(size_t)a * 3 + 1];
  float z = pos[(size_t)b * 3 + 2] - pos[(size_t)a * 3 + 2];
  float inv = 1.f / (sqrtf(x * x + y * y + z * z) + 1e-10f);
  out[(size_t)e * 3] = x * inv; out[(size_t)e * 3 + 1] = y * inv; out[(size_t)e * 3 + 2] = z * inv;
}

// ---------------------------------------------------------------------------
// Host orchestration
// ---------------------------------------------------------------------------
enum {
  IN_Z = 0, IN_TP, IN_POS, IN_BATCH, IN_EI,
  P_DD1W, P_DD1B, P_DD2W, P_DD2B,
  P_EE1W, P_EE1B, P_EE2W, P_EE2B,
  P_EF1W, P_EF1B, P_EF2W, P_EF2B,
  P_EM1W, P_EM1B, P_EM2W, P_EM2B, P_EM3W, P_EM3B,  // dead code: skipped
  P_EP1W, P_EP1B, P_EP2W, P_EP2B, P_EP3W, P_EP3B,
  P_HUPDW, P_HUPDB,
  P_LATW, P_LATB,
  P_MSGW, P_MSGB,
  P_ND1W, P_ND1B, P_ND2W, P_ND2B, P_NDBNG, P_NDBNB,
  P_NN1W, P_NN1B, P_NN2W, P_NN2B, P_NNBNG, P_NNBNB,
  P_NT1W, P_NT1B, P_NT2W, P_NT2B, P_NTBNG, P_NTBNB,
  P_XWW, P_XWB
};

extern "C" void kernel_launch(void* const* d_in, const int* in_sizes, int n_in,
                              void* d_out, int out_size, void* d_ws, size_t ws_size,
                              hipStream_t stream) {
  (void)n_in; (void)out_size; (void)ws_size;
  const float* z   = (const float*)d_in[IN_Z];
  const float* tp  = (const float*)d_in[IN_TP];
  const float* pin = (const float*)d_in[IN_POS];
  const int* batch = (const int*)d_in[IN_BATCH];
  const int* ei    = (const int*)d_in[IN_EI];

  int B = in_sizes[IN_Z] / 32;
  int N = in_sizes[IN_BATCH];
  int E = in_sizes[IN_EI] / 2;
  const int* src = ei;
  const int* dst = ei + E;

  const float* F[64];
  for (int i = 0; i < 55; ++i) F[i] = (const float*)d_in[i];

  float* ws = (float*)d_ws;
  size_t off = 0;
  auto alloc = [&](size_t n) { float* p = ws + off; off += n; return p; };
  float* h        = alloc((size_t)B * 64);
  float* he       = alloc((size_t)N * 64);
  float* t128     = alloc((size_t)N * 128);   // nt1 out; reused as nd hidden
  float* bn_mean  = alloc(128);
  float* bn_var   = alloc(128);
  float* eattr    = alloc((size_t)E * 4);
  float* pos      = alloc((size_t)N * 3 + 4);
  float* agg      = alloc((size_t)N * 64);
  float* posdelta = alloc((size_t)N * 3 + 4);
  float* deg      = alloc(N);
  float* tmp64    = alloc((size_t)N * 64);
  float* nnA      = alloc((size_t)B * 64);
  _Float16* he16  = (_Float16*)alloc((size_t)N * 32);   // f16 mirror of he

  // f16 weight pool (~220 KB), all buffers padded to K%32==0, NC%16==0
  _Float16* wpool = (_Float16*)alloc(128 * 1024);
  size_t hoff = 0;
  auto halloc = [&](size_t n) { _Float16* p = wpool + hoff; hoff += (n + 7) & ~(size_t)7; return p; };
  _Float16* ee1h = halloc(128 * 64);
  _Float16* ee2h = halloc(64 * 16);
  _Float16* dd1h = halloc(128 * 64);
  _Float16* dd2h = halloc(64 * 16);
  _Float16* ef1h = halloc(128 * 64);
  _Float16* ef2h = halloc(64 * 16);
  _Float16* ep1h = halloc(128 * 128);
  _Float16* ep2h = halloc(128 * 128);
  _Float16* ep3h = halloc(128 * 16);
  _Float16* msgh = halloc(160 * 64);
  _Float16* huph = halloc(128 * 64);
  _Float16* nt1h = halloc(64 * 128);
  _Float16* nt2h = halloc(128 * 64);
  _Float16* nd1h = halloc(64 * 64);
  _Float16* nd2h = halloc(64 * 16);
  _Float16* nn1h = halloc(64 * 64);
  _Float16* nn2h = halloc(64 * 16);

  float* out = (float*)d_out;
  size_t off0 = 0;
  size_t off1 = off0 + (size_t)N * 5;   // distances
  size_t off2 = off1 + (size_t)E;       // directions
  size_t off3 = off2 + (size_t)E * 3;   // edge_feats
  size_t off4 = off3 + (size_t)E * 4;   // num_nodes
  size_t off5 = off4 + (size_t)B;       // edge_existence

  // --- one-shot weight conversion to padded f16 (loop-invariant) ---
  auto cvt = [&](const float* s_, _Float16* d_, int K_, int NC_, int KP_, int NCP_) {
    k_cvt_pad<<<(KP_ * NCP_ + 255) / 256, 256, 0, stream>>>(s_, d_, K_, NC_, KP_, NCP_);
  };
  cvt(F[P_EE1W], ee1h, 128, 64, 128, 64);   cvt(F[P_EE2W], ee2h, 64, 1, 64, 16);
  cvt(F[P_DD1W], dd1h, 128, 64, 128, 64);   cvt(F[P_DD2W], dd2h, 64, 1, 64, 16);
  cvt(F[P_EF1W], ef1h, 128, 64, 128, 64);   cvt(F[P_EF2W], ef2h, 64, 4, 64, 16);
  cvt(F[P_EP1W], ep1h, 128, 128, 128, 128); cvt(F[P_EP2W], ep2h, 128, 128, 128, 128);
  cvt(F[P_EP3W], ep3h, 128, 4, 128, 16);    cvt(F[P_MSGW], msgh, 133, 64, 160, 64);
  cvt(F[P_HUPDW], huph, 128, 64, 128, 64);  cvt(F[P_NT1W], nt1h, 64, 128, 64, 128);
  cvt(F[P_NT2W], nt2h, 128, 64, 128, 64);   cvt(F[P_ND1W], nd1h, 64, 64, 64, 64);
  cvt(F[P_ND2W], nd2h, 64, 5, 64, 16);      cvt(F[P_NN1W], nn1h, 64, 64, 64, 64);
  cvt(F[P_NN2W], nn2h, 64, 1, 64, 16);

  // h = [z, tp] @ lin_latent
  k_latent<<<(B * 64 + 255) / 256, 256, 0, stream>>>(z, tp, F[P_LATW], F[P_LATB], h, B);

  // node transform: he = nt2( relu(bn( nt1(h[batch]) )) )
  k_gemm_t<64, 0, 128, 128, 0, true><<<dim3(N / 16, 8), 32, 0, stream>>>(
      h, nullptr, batch, nt1h, F[P_NT1B], t128);
  k_bn_stats<<<128, 256, 0, stream>>>(t128, N, 128, 0, bn_mean, bn_var);
  k_bn_apply<<<(unsigned)(((size_t)N * 128 + 255) / 256), 256, 0, stream>>>(
      t128, N, 128, 0, 1, bn_mean, bn_var, F[P_NTBNG], F[P_NTBNB]);
  k_gemm_t<128, 0, 64, 64, 0, false><<<dim3(N / 16, 4), 32, 0, stream>>>(
      t128, nullptr, nullptr, nt2h, F[P_NT2B], he);
  k_cvt_f16<<<(N * 64 + 255) / 256, 256, 0, stream>>>(he, he16, N * 64);

  // node decode: softmax( nd2( bn(relu(nd1(he))) ) )
  k_gemm_t<64, 0, 64, 64, 0, false><<<dim3(N / 16, 4), 32, 0, stream>>>(
      he, nullptr, nullptr, nd1h, F[P_ND1B], t128);
  k_bn_stats<<<64, 256, 0, stream>>>(t128, N, 64, 1, bn_mean, bn_var);
  k_bn_apply<<<(unsigned)(((size_t)N * 64 + 255) / 256), 256, 0, stream>>>(
      t128, N, 64, 1, 0, bn_mean, bn_var, F[P_NDBNG], F[P_NDBNB]);
  k_gemm_t<64, 0, 5, 16, 0, false><<<dim3(N / 16, 1), 32, 0, stream>>>(
      t128, nullptr, nullptr, nd2h, F[P_ND2B], out + off0);
  k_softmax_rows<<<(N + 255) / 256, 256, 0, stream>>>(out + off0, N, 5);

  // fused edge heads (ee/dd/ef/ep) -- em* MLP is dead in reference, skipped
  k_edge_heads<<<E / 16, 32, 0, stream>>>(
      he16, src, dst,
      ee1h, F[P_EE1B], ee2h, F[P_EE2B],
      dd1h, F[P_DD1B], dd2h, F[P_DD2B],
      ef1h, F[P_EF1B], ef2h, F[P_EF2B],
      ep1h, F[P_EP1B], ep2h, F[P_EP2B],
      ep3h, F[P_EP3B],
      out + off5, out + off1, out + off3, eattr, E);

  // geometry init
  k_posnorm<<<(N + 255) / 256, 256, 0, stream>>>(pin, pos, N);
  k_fill<<<(N + 255) / 256, 256, 0, stream>>>(deg, (size_t)N, 0.f);
  k_deg<<<(E + 255) / 256, 256, 0, stream>>>(dst, deg, E);

  // EGNN refinement (6 steps)
  for (int it = 0; it < 6; ++it) {
    k_fill<<<(unsigned)(((size_t)N * 64 + 255) / 256), 256, 0, stream>>>(agg, (size_t)N * 64, 0.f);
    k_fill<<<(unsigned)(((size_t)N * 3 + 255) / 256), 256, 0, stream>>>(posdelta, (size_t)N * 3, 0.f);
    k_egnn_edge<<<E / 16, 32, 0, stream>>>(he16, pos, src, dst, eattr,
                                           msgh, F[P_MSGB], F[P_XWW], F[P_XWB],
                                           agg, posdelta, E);
    k_gemm_t<64, 64, 64, 64, 0, false><<<dim3(N / 16, 4), 32, 0, stream>>>(
        he, agg, nullptr, huph, F[P_HUPDB], tmp64);
    k_axpy_dual<<<(unsigned)(((size_t)N * 64 + 255) / 256), 256, 0, stream>>>(
        he, he16, tmp64, (size_t)N * 64);
    float coef = 0.5f / (float)(it + 1);
    k_pos_update<<<(N + 255) / 256, 256, 0, stream>>>(pos, posdelta, deg, coef, N);
  }

  // directions
  k_directions<<<(E + 255) / 256, 256, 0, stream>>>(pos, src, dst, out + off2, E);

  // num_nodes: relu( nn2( bn(relu(nn1(h))) ) )
  k_gemm_t<64, 0, 64, 64, 1, false><<<dim3(B / 16, 4), 32, 0, stream>>>(
      h, nullptr, nullptr, nn1h, F[P_NN1B], nnA);
  k_bn_stats<<<64, 256, 0, stream>>>(nnA, B, 64, 0, bn_mean, bn_var);
  k_bn_apply<<<(unsigned)(((size_t)B * 64 + 255) / 256), 256, 0, stream>>>(
      nnA, B, 64, 0, 0, bn_mean, bn_var, F[P_NNBNG], F[P_NNBNB]);
  k_gemm_t<64, 0, 1, 16, 1, false><<<dim3(B / 16, 1), 32, 0, stream>>>(
      nnA, nullptr, nullptr, nn2h, F[P_NN2B], out + off4);
}
